// MLPForwardPolicy_74758200754580
// MI455X (gfx1250) — matrix operand order repaired and checked by reading, not compiled, not run on hardware
//
#include <hip/hip_runtime.h>
#include <hip/hip_bf16.h>

typedef __attribute__((ext_vector_type(16))) _Float16 h16;
typedef __attribute__((ext_vector_type(8)))  _Float16 h8;
typedef __attribute__((ext_vector_type(4)))  _Float16 h4;
typedef __attribute__((ext_vector_type(8)))  float    f8;
typedef __attribute__((ext_vector_type(4)))  float    f4;

// ---------------------------------------------------------------------------
// fp32 -> f16 conversion (activations only; 16 MB read, 8 MB write)
// ---------------------------------------------------------------------------
__global__ __launch_bounds__(256) void cvt_f32_f16(const float* __restrict__ in,
                                                   _Float16* __restrict__ out,
                                                   int n4) {
    int i = blockIdx.x * blockDim.x + threadIdx.x;
    if (i < n4) {
        f4 v = *(const f4*)(in + 4 * i);
        h4 o;
        o.x = (_Float16)v.x; o.y = (_Float16)v.y;
        o.z = (_Float16)v.z; o.w = (_Float16)v.w;
        *(h4*)(out + 4 * i) = o;
    }
}

// ---------------------------------------------------------------------------
// Weight-streaming WMMA GEMM:  out[256,N] = act(A[256,K] * W[K,N] + bias)
//   - A: f16 row-major (lda), hot in L2 (RT loads)
//   - W: fp32 row-major (ldw), streamed once with NT hint, cvt to f16 in regs,
//     staged into LDS in WMMA-B fragment order (double-buffered, BK=64)
//   - 8 waves; wave w owns rows [32w,32w+32); block owns cols [32bx,32bx+32)
//   - 8x v_wmma_f32_16x16x32_f16 per barrier
// ---------------------------------------------------------------------------
__global__ __launch_bounds__(256)
void gemm_wmma_f16(const _Float16* __restrict__ A, const float* __restrict__ W,
                   const float* __restrict__ bias, void* __restrict__ outp,
                   int K, int lda, int ldw, int ldo, int relu, int f16out) {
    // 2 buffers * (2 k-chunks * 2 n-tiles * 32 lanes * 16 halves) = 8 KB
    __shared__ __attribute__((aligned(32))) _Float16 lsB[2][2048];

    const int tid  = threadIdx.x;
    const int lane = tid & 31;
    const int n0   = blockIdx.x * 32;
    const int mrow = (tid >> 5) * 32;

    // --- staging map: thread owns 2 groups of 4 consecutive-N elements of the
    //     64K x 32N fp32 tile; each group = one global_load_b128
    long woff[2]; int soff[2];
#pragma unroll
    for (int v = 0; v < 2; ++v) {
        int e0 = (tid + 256 * v) * 4;
        int k = e0 >> 5, n = e0 & 31;        // n multiple of 4, fixed k per group
        woff[v] = (long)k * ldw + n;
        int c = k >> 5, kc = k & 31;
        int lanew = ((kc & 8) ? 16 : 0) + (n & 15);  // lanes 16-31 hold K=8..15,24..31
        soff[v] = c * 1024 + ((n >> 4) * 32 + lanew) * 16 + (kc & 7) + ((kc & 16) >> 1);
    }

    // --- A fragment addressing (ISA 16-bit A 16x32 layout)
    const int am  = lane & 15;
    const int akb = (lane >> 4) * 8;
    const _Float16* a0p = A + (long)(mrow + am) * lda + akb;
    const _Float16* a1p = a0p + 16 * (long)lda;

    const float* Wp = W + n0;

    f8 acc00 = {}, acc01 = {}, acc10 = {}, acc11 = {};

    const int nIter = K >> 6;   // K % 64 == 0 for all layers (16384/9216/8960)

    // prologue: stage tile 0 into buffer 0
#pragma unroll
    for (int v = 0; v < 2; ++v) {
        f4 wv = __builtin_nontemporal_load((const f4*)(Wp + woff[v]));
        _Float16* d = &lsB[0][soff[v]];
        d[0]  = (_Float16)wv.x; d[16] = (_Float16)wv.y;
        d[32] = (_Float16)wv.z; d[48] = (_Float16)wv.w;
    }
    __syncthreads();

    int buf = 0;
    for (int it = 0; it < nIter; ++it) {
        const int k0 = it << 6;
        const bool more = (it + 1) < nIter;

        // issue next W tile's global loads early (latency hidden by WMMAs)
        f4 wv[2];
        if (more) {
            const float* Wn = Wp + (long)(k0 + 64) * ldw;
#pragma unroll
            for (int v = 0; v < 2; ++v)
                wv[v] = __builtin_nontemporal_load((const f4*)(Wn + woff[v]));
        }

        // compute on current buffer: 2 k-chunks x (2x2) WMMA tiles
#pragma unroll
        for (int c = 0; c < 2; ++c) {
            const int kc = k0 + c * 32;
            union { h16 v; h8 h[2]; } a0, a1, b0, b1;
            a0.h[0] = *(const h8*)(a0p + kc);
            a0.h[1] = *(const h8*)(a0p + kc + 16);
            a1.h[0] = *(const h8*)(a1p + kc);
            a1.h[1] = *(const h8*)(a1p + kc + 16);
            const _Float16* bb = &lsB[buf][c * 1024];
            b0.h[0] = *(const h8*)(bb + lane * 16);
            b0.h[1] = *(const h8*)(bb + lane * 16 + 8);
            b1.h[0] = *(const h8*)(bb + 512 + lane * 16);
            b1.h[1] = *(const h8*)(bb + 512 + lane * 16 + 8);

            acc00 = __builtin_amdgcn_wmma_f32_16x16x32_f16(false, a0.v, false, b0.v, (short)0, acc00, false, false);
            acc01 = __builtin_amdgcn_wmma_f32_16x16x32_f16(false, a0.v, false, b1.v, (short)0, acc01, false, false);
            acc10 = __builtin_amdgcn_wmma_f32_16x16x32_f16(false, a1.v, false, b0.v, (short)0, acc10, false, false);
            acc11 = __builtin_amdgcn_wmma_f32_16x16x32_f16(false, a1.v, false, b1.v, (short)0, acc11, false, false);
        }

        // convert + store next tile into the other buffer
        if (more) {
            _Float16* dst = &lsB[buf ^ 1][0];
#pragma unroll
            for (int v = 0; v < 2; ++v) {
                _Float16* d = dst + soff[v];
                d[0]  = (_Float16)wv[v].x; d[16] = (_Float16)wv[v].y;
                d[32] = (_Float16)wv[v].z; d[48] = (_Float16)wv[v].w;
            }
        }
        buf ^= 1;
        __syncthreads();
    }

    // epilogue (ISA 32-bit C/D layout: lanes 0-15 -> M=r, lanes 16-31 -> M=8+r)
    const int em = (lane >> 4) * 8;
    const int en = lane & 15;
    f8 accs[2][2] = {{acc00, acc01}, {acc10, acc11}};
#pragma unroll
    for (int i = 0; i < 2; ++i)
#pragma unroll
        for (int j = 0; j < 2; ++j) {
            f8 a = accs[i][j];
            int col = n0 + j * 16 + en;
            float bv = bias[col];
#pragma unroll
            for (int r = 0; r < 8; ++r) {
                int row = mrow + i * 16 + em + r;
                float v = a[r] + bv;
                if (relu) v = v > 0.f ? v : 0.f;
                if (f16out) ((_Float16*)outp)[(long)row * ldo + col] = (_Float16)v;
                else        ((float*)outp)[(long)row * ldo + col] = v;
            }
        }
}

// ---------------------------------------------------------------------------
// Selection branch: one block per batch row (256 threads).
//   y = h @ Ws + bs ; p = softmax(y) ; v = mask ? -1e9 : p ; p2 = softmax(v)
//   score = mask ? -inf : log(p2) + gumbel ; idx = argmax (first on ties)
// ---------------------------------------------------------------------------
__global__ __launch_bounds__(256)
void select_kernel(const _Float16* __restrict__ H3, const float* __restrict__ Ws,
                   const float* __restrict__ bs, const int* __restrict__ mask,
                   const float* __restrict__ u, float* __restrict__ selprob,
                   float* __restrict__ coord) {
    __shared__ float sh[256];
    __shared__ float hrow[256];
    __shared__ int   si[256];
    __shared__ float sp[256];
    const int row = blockIdx.x, j = threadIdx.x;

    hrow[j] = (float)H3[(long)row * 9216 + 8960 + j];
    __syncthreads();

    float acc = bs[j];
#pragma unroll 4
    for (int k = 0; k < 256; ++k) acc += hrow[k] * Ws[k * 256 + j];

    // softmax #1
    sh[j] = acc; __syncthreads();
    for (int s = 128; s > 0; s >>= 1) { if (j < s) sh[j] = fmaxf(sh[j], sh[j + s]); __syncthreads(); }
    float mx = sh[0]; __syncthreads();
    float e = expf(acc - mx);
    sh[j] = e; __syncthreads();
    for (int s = 128; s > 0; s >>= 1) { if (j < s) sh[j] += sh[j + s]; __syncthreads(); }
    float p = e / sh[0]; __syncthreads();

    int m = mask[row * 256 + j];
    float v = m ? -1e9f : p;

    // softmax #2
    sh[j] = v; __syncthreads();
    for (int s = 128; s > 0; s >>= 1) { if (j < s) sh[j] = fmaxf(sh[j], sh[j + s]); __syncthreads(); }
    float mx2 = sh[0]; __syncthreads();
    float e2 = expf(v - mx2);
    sh[j] = e2; __syncthreads();
    for (int s = 128; s > 0; s >>= 1) { if (j < s) sh[j] += sh[j + s]; __syncthreads(); }
    float p2 = e2 / sh[0]; __syncthreads();

    // gumbel-max
    float g = -logf(-logf(u[row * 256 + j]));
    float score = m ? -__builtin_inff() : (logf(p2) + g);

    sh[j] = score; si[j] = j; sp[j] = p2; __syncthreads();
    for (int s = 128; s > 0; s >>= 1) {
        if (j < s) {
            float o = sh[j + s]; int oi = si[j + s];
            if (o > sh[j] || (o == sh[j] && oi < si[j])) { sh[j] = o; si[j] = oi; }
        }
        __syncthreads();
    }
    if (j == 0) {
        int idx = si[0];
        selprob[row] = sp[idx];
        coord[row * 2 + 0] = (float)(idx >> 4);  // idx // 16
        coord[row * 2 + 1] = (float)(idx & 15);  // idx % 16
    }
}

// ---------------------------------------------------------------------------
// Action head epilogue: softmax over 64 logits per row, scaled by sel_prob
// ---------------------------------------------------------------------------
__global__ __launch_bounds__(64)
void action_final(const float* __restrict__ logits, const float* __restrict__ selprob,
                  float* __restrict__ out) {
    __shared__ float sh[64];
    const int row = blockIdx.x, j = threadIdx.x;
    float x = logits[row * 64 + j];
    sh[j] = x; __syncthreads();
    for (int s = 32; s > 0; s >>= 1) { if (j < s) sh[j] = fmaxf(sh[j], sh[j + s]); __syncthreads(); }
    float mx = sh[0]; __syncthreads();
    float e = expf(x - mx);
    sh[j] = e; __syncthreads();
    for (int s = 32; s > 0; s >>= 1) { if (j < s) sh[j] += sh[j + s]; __syncthreads(); }
    out[row * 64 + j] = (e / sh[0]) * selprob[row];
}

// ---------------------------------------------------------------------------
extern "C" void kernel_launch(void* const* d_in, const int* in_sizes, int n_in,
                              void* d_out, int out_size, void* d_ws, size_t ws_size,
                              hipStream_t stream) {
    const float* s    = (const float*)d_in[0];
    const int*   mask = (const int*)d_in[1];
    const float* u    = (const float*)d_in[2];
    const float* W1   = (const float*)d_in[3];
    const float* b1   = (const float*)d_in[4];
    const float* W2   = (const float*)d_in[5];
    const float* b2   = (const float*)d_in[6];
    const float* W3   = (const float*)d_in[7];
    const float* b3   = (const float*)d_in[8];
    const float* Wa   = (const float*)d_in[9];
    const float* ba   = (const float*)d_in[10];
    const float* Ws   = (const float*)d_in[11];
    const float* bs   = (const float*)d_in[12];

    char* ws = (char*)d_ws;
    _Float16* X16     = (_Float16*)(ws);                         // 256*16384*2 = 8388608
    _Float16* H1      = (_Float16*)(ws + 8388608);               // 256*9216*2  = 4718592
    _Float16* H2      = (_Float16*)(ws + 13107200);
    _Float16* H3      = (_Float16*)(ws + 17825792);
    float*    logitsA = (float*)   (ws + 22544384);              // 256*64*4
    float*    selprob = (float*)   (ws + 22609920);              // 256*4

    float* out = (float*)d_out;  // [0,16384): final_action_prob ; [16384,16896): coordinates

    // x = s.reshape(B, -1) -> f16
    cvt_f32_f16<<<(256 * 16384 / 4 + 255) / 256, 256, 0, stream>>>(s, X16, 256 * 16384 / 4);

    // three MLP layers (relu, f16 activations out)
    gemm_wmma_f16<<<288, 256, 0, stream>>>(X16, W1, b1, H1, 16384, 16384, 9216, 9216, 1, 1);
    gemm_wmma_f16<<<288, 256, 0, stream>>>(H1,  W2, b2, H2,  9216,  9216, 9216, 9216, 1, 1);
    gemm_wmma_f16<<<288, 256, 0, stream>>>(H2,  W3, b3, H3,  9216,  9216, 9216, 9216, 1, 1);

    // action head logits: x[:, :8960] @ Wa + ba (no relu, f32 out)
    gemm_wmma_f16<<<2, 256, 0, stream>>>(H3, Wa, ba, logitsA, 8960, 9216, 64, 64, 0, 0);

    // selection branch -> sel_prob + coordinates
    select_kernel<<<256, 256, 0, stream>>>(H3, Ws, bs, mask, u, selprob, out + 256 * 64);

    // final_action_prob = softmax(logits) * sel_prob
    action_final<<<256, 64, 0, stream>>>(logitsA, selprob, out);
}
